// LSTMModel_84653805404857
// MI455X (gfx1250) — matrix operand order repaired and checked by reading, not compiled, not run on hardware
//
#include <hip/hip_runtime.h>
#include <math.h>

// ---------------------------------------------------------------------------
// Problem constants (from reference): B=64, T=512, I=128, H=512, L=2, O=1
// ---------------------------------------------------------------------------
#define BATCH   64
#define TSTEPS  512
#define IDIM    128
#define HDIM    512
#define NWG     16          // workgroups in the persistent recurrent kernel
#define G4H     2048        // 4*H gate rows
#define WPAD    8           // bf16 elements of LDS row padding (16B, bank skew)

typedef __attribute__((ext_vector_type(16))) __bf16 v16bf;
typedef __attribute__((ext_vector_type(8)))  float  v8f;

struct alignas(16) U4 { unsigned x, y, z, w; };
struct U8 { U4 lo, hi; };

__device__ __forceinline__ unsigned short f32_to_bf16(float f) {
  unsigned u = __float_as_uint(f);
  u += 0x7FFFu + ((u >> 16) & 1u);      // round-to-nearest-even
  return (unsigned short)(u >> 16);
}

// A fragment (16x32 MxK bf16) from a per-lane base pointer:
// lo 16B = K..K+7, hi 16B at +16 elements = K+16..K+23 (ISA A layout).
__device__ __forceinline__ v16bf frag_a(const unsigned short* p) {
  U8 u;
  u.lo = *(const U4*)p;
  u.hi = *(const U4*)(p + 16);
  return __builtin_bit_cast(v16bf, u);
}

// B fragment (32x16 KxN bf16) from a per-lane LDS base pointer:
// lo 16B = K..K+7, hi 16B at +16 elements = K+16..K+23 (matches A layout).
__device__ __forceinline__ v16bf frag_b(const unsigned short* p) {
  U8 u;
  u.lo = *(const U4*)p;
  u.hi = *(const U4*)(p + 16);
  return __builtin_bit_cast(v16bf, u);
}

// Software grid barrier across the NWG resident workgroups (agent scope).
__device__ __forceinline__ void grid_barrier(unsigned* cnt, unsigned* gen) {
  __threadfence();               // release our h stores to device scope
  __syncthreads();
  if (threadIdx.x == 0) {
    unsigned g = __hip_atomic_load(gen, __ATOMIC_ACQUIRE, __HIP_MEMORY_SCOPE_AGENT);
    unsigned a = __hip_atomic_fetch_add(cnt, 1u, __ATOMIC_ACQ_REL, __HIP_MEMORY_SCOPE_AGENT);
    if (a == NWG - 1) {
      __hip_atomic_store(cnt, 0u, __ATOMIC_RELAXED, __HIP_MEMORY_SCOPE_AGENT);
      __hip_atomic_fetch_add(gen, 1u, __ATOMIC_ACQ_REL, __HIP_MEMORY_SCOPE_AGENT);
    } else {
      while (__hip_atomic_load(gen, __ATOMIC_ACQUIRE, __HIP_MEMORY_SCOPE_AGENT) == g) {
        __builtin_amdgcn_s_sleep(2);
      }
    }
  }
  __syncthreads();
  __threadfence();               // acquire: invalidate stale WGP$ lines
}

// ---------------------------------------------------------------------------
// Prep kernels
// ---------------------------------------------------------------------------
__global__ void init_bar_kernel(unsigned* bar) {
  if (threadIdx.x < 8) bar[threadIdx.x] = 0u;
}

// x [B,T,I] fp32  ->  xt [T,B,I] bf16 (time-major for per-step A loads)
__global__ void convert_x_kernel(const float* __restrict__ x,
                                 unsigned short* __restrict__ xt) {
  const size_t total = (size_t)BATCH * TSTEPS * IDIM;
  for (size_t idx = (size_t)blockIdx.x * blockDim.x + threadIdx.x; idx < total;
       idx += (size_t)gridDim.x * blockDim.x) {
    size_t b = idx / ((size_t)TSTEPS * IDIM);
    size_t r = idx - b * (size_t)TSTEPS * IDIM;
    size_t t = r / IDIM;
    size_t i = r - t * IDIM;
    xt[t * (size_t)BATCH * IDIM + b * IDIM + i] = f32_to_bf16(x[idx]);
  }
}

// Pack [w_ih | w_hh] into per-WG WMMA-friendly bf16 layout:
//   Wp[wg*128 + n][k],  n in [0,128): gate = n>>5, j = wg*32 + (n&31),
//   original gate row r = gate*512 + j;  k<Kin -> w_ih[r][k], else w_hh[r][k-Kin].
// Also biasp[row] = b_ih[r] + b_hh[r] (packed order).
__global__ void pack_w_kernel(const float* __restrict__ w_ih,
                              const float* __restrict__ w_hh,
                              const float* __restrict__ b_ih,
                              const float* __restrict__ b_hh,
                              unsigned short* __restrict__ Wp,
                              float* __restrict__ biasp, int Kin) {
  const int Ktot = Kin + HDIM;
  const size_t total = (size_t)G4H * Ktot;
  for (size_t idx = (size_t)blockIdx.x * blockDim.x + threadIdx.x; idx < total;
       idx += (size_t)gridDim.x * blockDim.x) {
    int row = (int)(idx / Ktot);
    int k   = (int)(idx - (size_t)row * Ktot);
    int wg   = row >> 7;
    int n    = row & 127;
    int gate = n >> 5;
    int j    = (wg << 5) + (n & 31);
    int r    = gate * HDIM + j;
    float v = (k < Kin) ? w_ih[(size_t)r * Kin + k]
                        : w_hh[(size_t)r * HDIM + (k - Kin)];
    Wp[idx] = f32_to_bf16(v);
    if (k == 0) biasp[row] = b_ih[r] + b_hh[r];
  }
}

// ---------------------------------------------------------------------------
// Persistent fused LSTM layer: 16 WGs x 256 threads, one grid barrier / step.
// Weights for this WG's 128 gate columns live in LDS for the whole kernel.
// Per step GEMM: [x_t | h_cur] (64 x Ktot) @ Wlds^T (Ktot x 128) via bf16 WMMA,
// with both A (global) and B (LDS) fragments double-buffered in registers.
// KIN is compile-time so the k-loop unrolls with register renaming (no movs).
// Dynamic LDS layout: wlds[128][KP] bf16 | gates[64][128] f32 | creg[64][32] f32
// ---------------------------------------------------------------------------
template <int KIN, bool WSEQ>
__global__ void __launch_bounds__(256)
lstm_layer_kernel(const unsigned short* __restrict__ Xseq,
                  const unsigned short* __restrict__ Wp,
                  const float* __restrict__ biasp,
                  const float* __restrict__ h0,   // [64*512] fp32 (hx[layer])
                  const float* __restrict__ c0,   // [64*512] fp32 (cx[layer])
                  unsigned short* __restrict__ hbuf,  // [2][64*512] bf16
                  unsigned short* __restrict__ Hseq,  // [T][64][512] bf16 (WSEQ)
                  float* __restrict__ hn_out, float* __restrict__ cn_out,
                  unsigned* __restrict__ bar) {
  constexpr int Ktot = KIN + HDIM;
  constexpr int KP   = Ktot + WPAD;           // padded LDS row stride (elements)
  constexpr int ktx  = KIN >> 5;              // k-tiles from x_t
  constexpr int ktn  = ktx + (HDIM >> 5);     // total k-tiles

  extern __shared__ char smem[];
  unsigned short* wlds = (unsigned short*)smem;                 // 128 * KP bf16
  float* gates = (float*)(smem + (size_t)128 * KP * 2);         // 64 * 128 f32
  float* creg  = gates + BATCH * 128;                           // 64 * 32  f32

  const int tid  = threadIdx.x;
  const int lane = tid & 31;
  const int w    = tid >> 5;            // wave 0..7
  const int wg   = blockIdx.x;          // 0..15
  const int m0   = (w & 3) * 16;        // batch tile
  const int nb   = (w >> 2);            // 0..1 -> 64-column half of WG slice
  const int wrow = wg * 128;            // packed weight row base for this WG

  // ---- Stage this WG's weight slice into LDS (once, reused all 512 steps) ----
  constexpr int c8max = Ktot >> 3;      // 16B chunks per row
  for (int idx = tid; idx < 128 * c8max; idx += 256) {
    int row = idx / c8max;
    int c8  = idx - row * c8max;
    U4 v = *(const U4*)(Wp + (size_t)(wrow + row) * Ktot + c8 * 8);
    *(U4*)(wlds + (size_t)row * KP + c8 * 8) = v;
  }

  // Per-wave bias registers: C/D layout is lane->N, so one value per N-tile.
  float biasv[4];
  #pragma unroll
  for (int nt = 0; nt < 4; ++nt)
    biasv[nt] = biasp[wrow + nb * 64 + nt * 16 + (lane & 15)];

  // Per-lane LDS base addresses for the 4 B tiles (hoisted out of all loops).
  const unsigned short* b_base[4];
  #pragma unroll
  for (int nt = 0; nt < 4; ++nt)
    b_base[nt] = wlds + (size_t)(nb * 64 + nt * 16 + (lane & 15)) * KP +
                 ((lane >> 4) << 3);

  // Init: cell state slice from cx; seed h double-buffer[0] from hx.
  for (int idx = tid; idx < BATCH * 32; idx += 256) {
    int m = idx >> 5, j = idx & 31, col = wg * 32 + j;
    creg[m * 32 + j] = c0[m * HDIM + col];
    hbuf[m * HDIM + col] = f32_to_bf16(h0[m * HDIM + col]);
  }
  grid_barrier(bar, bar + 1);           // also covers the LDS staging barrier

  const int a_lane_row = m0 + (lane & 15);
  const int a_lane_k   = (lane >> 4) << 3;

  for (int t = 0; t < TSTEPS; ++t) {
    v8f acc[4];
    #pragma unroll
    for (int i = 0; i < 4; ++i)
      #pragma unroll
      for (int j = 0; j < 8; ++j) acc[i][j] = biasv[i];   // bias pre-folded

    const unsigned short* xt = Xseq + (size_t)t * BATCH * KIN;
    const unsigned short* hc = hbuf + (size_t)(t & 1) * BATCH * HDIM;

    // Per-lane A base pointers for the two K segments; hm is pre-shifted so
    // both segments address as base + kt*32.
    const unsigned short* xa = xt + (size_t)a_lane_row * KIN + a_lane_k;
    const unsigned short* hm = hc + (size_t)a_lane_row * HDIM + a_lane_k -
                               (size_t)ktx * 32;

    // Pipelined prologue: A and all 4 B fragments of tile 0.
    v16bf a = frag_a(xa);
    v16bf bq[4];
    #pragma unroll
    for (int nt = 0; nt < 4; ++nt) bq[nt] = frag_b(b_base[nt]);

    #pragma unroll 4
    for (int kt = 0; kt < ktn; ++kt) {
      const bool more = (kt + 1 < ktn);
      v16bf an;
      v16bf bn[4];
      if (more) {
        const int kn = kt + 1;
        const unsigned short* ap = (kn < ktx ? xa : hm) + (size_t)kn * 32;
        an = frag_a(ap);
        #pragma unroll
        for (int nt = 0; nt < 4; ++nt)
          bn[nt] = frag_b(b_base[nt] + (size_t)kn * 32);
      }
      #pragma unroll
      for (int nt = 0; nt < 4; ++nt)
        acc[nt] = __builtin_amdgcn_wmma_f32_16x16x32_bf16(
            false, a, false, bq[nt], (short)0, acc[nt], false, false);
      if (more) {
        a = an;
        #pragma unroll
        for (int nt = 0; nt < 4; ++nt) bq[nt] = bn[nt];
      }
    }

    // Scatter D tiles -> gates LDS. D layout: lane -> N, vgpr r -> M.
    #pragma unroll
    for (int nt = 0; nt < 4; ++nt) {
      int n  = nb * 64 + nt * 16 + (lane & 15);
      int mb = m0 + ((lane >> 4) << 3);
      #pragma unroll
      for (int r = 0; r < 8; ++r) gates[(mb + r) * 128 + n] = acc[nt][r];
    }
    __syncthreads();

    // LSTM cell update for this WG's 64x32 slice (fp32 math, bias already in).
    unsigned short* hnext = hbuf + (size_t)((t + 1) & 1) * BATCH * HDIM;
    for (int idx = tid; idx < BATCH * 32; idx += 256) {
      int m = idx >> 5, j = idx & 31, col = wg * 32 + j;
      float gi = gates[m * 128 + j];
      float gf = gates[m * 128 + 32 + j];
      float gg = gates[m * 128 + 64 + j];
      float go = gates[m * 128 + 96 + j];
      float gate_i = 1.0f / (1.0f + expf(-gi));
      float gate_f = 1.0f / (1.0f + expf(-gf));
      float gate_g = tanhf(gg);
      float gate_o = 1.0f / (1.0f + expf(-go));
      float c = gate_f * creg[m * 32 + j] + gate_i * gate_g;
      creg[m * 32 + j] = c;
      float h = gate_o * tanhf(c);
      unsigned short hb = f32_to_bf16(h);
      hnext[m * HDIM + col] = hb;
      if (WSEQ) Hseq[(size_t)t * BATCH * HDIM + m * HDIM + col] = hb;
      if (t == TSTEPS - 1) {
        hn_out[m * HDIM + col] = h;
        cn_out[m * HDIM + col] = c;
      }
    }
    grid_barrier(bar, bar + 1);
  }
}

// Final FC: out[b] = dot(h1_T[b], fc_w) + fc_b   (64 x 512 @ 512 x 1)
__global__ void fc_kernel(const float* __restrict__ h1T,
                          const float* __restrict__ fc_w,
                          const float* __restrict__ fc_b,
                          float* __restrict__ out) {
  __shared__ float red[128];
  int b = blockIdx.x;
  float s = 0.0f;
  for (int k = threadIdx.x; k < HDIM; k += blockDim.x)
    s += h1T[b * HDIM + k] * fc_w[k];
  red[threadIdx.x] = s;
  __syncthreads();
  for (int off = 64; off > 0; off >>= 1) {
    if (threadIdx.x < off) red[threadIdx.x] += red[threadIdx.x + off];
    __syncthreads();
  }
  if (threadIdx.x == 0) out[b] = red[0] + fc_b[0];
}

// ---------------------------------------------------------------------------
// Launch
// ---------------------------------------------------------------------------
extern "C" void kernel_launch(void* const* d_in, const int* in_sizes, int n_in,
                              void* d_out, int out_size, void* d_ws, size_t ws_size,
                              hipStream_t stream) {
  (void)in_sizes; (void)n_in; (void)out_size; (void)ws_size;

  const float* x      = (const float*)d_in[0];
  const float* hx     = (const float*)d_in[1];   // [2,64,512]
  const float* cx     = (const float*)d_in[2];
  const float* w_ih_0 = (const float*)d_in[3];
  const float* w_hh_0 = (const float*)d_in[4];
  const float* b_ih_0 = (const float*)d_in[5];
  const float* b_hh_0 = (const float*)d_in[6];
  const float* w_ih_1 = (const float*)d_in[7];
  const float* w_hh_1 = (const float*)d_in[8];
  const float* b_ih_1 = (const float*)d_in[9];
  const float* b_hh_1 = (const float*)d_in[10];
  const float* fc_w   = (const float*)d_in[11];
  const float* fc_b   = (const float*)d_in[12];

  // Output layout: out[64] | hn[2*64*512] | cn[2*64*512]
  float* out = (float*)d_out;
  float* hn0 = out + 64;
  float* hn1 = hn0 + BATCH * HDIM;
  float* cn0 = out + 64 + 2 * BATCH * HDIM;
  float* cn1 = cn0 + BATCH * HDIM;

  // Workspace carve-up (256B aligned)
  char* ws = (char*)d_ws;
  size_t off = 0;
  auto take = [&](size_t bytes) {
    char* p = ws + off;
    off = (off + bytes + 255) & ~(size_t)255;
    return p;
  };
  unsigned short* xt     = (unsigned short*)take((size_t)TSTEPS * BATCH * IDIM * 2);
  unsigned short* h0seq  = (unsigned short*)take((size_t)TSTEPS * BATCH * HDIM * 2);
  unsigned short* W0p    = (unsigned short*)take((size_t)G4H * (IDIM + HDIM) * 2);
  unsigned short* W1p    = (unsigned short*)take((size_t)G4H * (HDIM + HDIM) * 2);
  float*          biasp0 = (float*)take((size_t)G4H * 4);
  float*          biasp1 = (float*)take((size_t)G4H * 4);
  unsigned short* hbuf0  = (unsigned short*)take((size_t)2 * BATCH * HDIM * 2);
  unsigned short* hbuf1  = (unsigned short*)take((size_t)2 * BATCH * HDIM * 2);
  unsigned*       bar    = (unsigned*)take(64);   // bar[0..1]=layer0, bar[2..3]=layer1

  init_bar_kernel<<<1, 64, 0, stream>>>(bar);
  convert_x_kernel<<<512, 256, 0, stream>>>(x, xt);
  pack_w_kernel<<<512, 256, 0, stream>>>(w_ih_0, w_hh_0, b_ih_0, b_hh_0, W0p, biasp0, IDIM);
  pack_w_kernel<<<512, 256, 0, stream>>>(w_ih_1, w_hh_1, b_ih_1, b_hh_1, W1p, biasp1, HDIM);

  // Dynamic LDS: weights (padded) + gates + cell slice.
  size_t shmem0 = (size_t)128 * (IDIM + HDIM + WPAD) * 2 +
                  (size_t)BATCH * 128 * 4 + (size_t)BATCH * 32 * 4;   // ~203 KB
  size_t shmem1 = (size_t)128 * (HDIM + HDIM + WPAD) * 2 +
                  (size_t)BATCH * 128 * 4 + (size_t)BATCH * 32 * 4;   // ~298 KB

  // Layer 0: X = x (K=128), writes full hidden sequence for layer 1.
  lstm_layer_kernel<IDIM, true><<<NWG, 256, shmem0, stream>>>(
      xt, W0p, biasp0, hx, cx, hbuf0, h0seq, hn0, cn0, bar);
  // Layer 1: X = h0seq (K=512), no sequence output needed.
  lstm_layer_kernel<HDIM, false><<<NWG, 256, shmem1, stream>>>(
      h0seq, W1p, biasp1, hx + BATCH * HDIM, cx + BATCH * HDIM,
      hbuf1, (unsigned short*)nullptr, hn1, cn1, bar + 2);
  fc_kernel<<<64, 128, 0, stream>>>(hn1, fc_w, fc_b, out);
}
